// MSSSIM_30116310680104
// MI455X (gfx1250) — matrix operand order repaired and checked by reading, not compiled, not run on hardware
//
#include <hip/hip_runtime.h>

// ---------------------------------------------------------------------------
// MS-SSIM on gfx1250: separable 11-tap Gaussian conv mapped onto
// V_WMMA_F32_16X16X32_F16 (Toeplitz matmul per 16x16 tile), f32 accumulate.
// ---------------------------------------------------------------------------

typedef __attribute__((ext_vector_type(16))) _Float16 v16h;
typedef __attribute__((ext_vector_type(8)))  _Float16 v8h;
typedef __attribute__((ext_vector_type(8)))  float    v8f;

// Toeplitz entry: out[n] = sum_t g[t] * in[n + t]  =>  T[k][n] = g[k - n].
// Gaussian computed analytically (one v_exp_f32, no memory): sigma=1.5,
// 1/sum(exp terms) = 0.26601329.
__device__ __forceinline__ float toep(int k, int n) {
    const int t = k - n;
    const float d = (float)(t - 5);
    const float g = __expf(d * d * (-1.0f / 4.5f)) * 0.26601329f;
    return ((unsigned)t < 11u) ? g : 0.0f;
}

// One wave = one 16x16 output tile of one (b,c) image.
// grid: (ntilesX*ntilesY, B*C), block: 32.
__global__ __launch_bounds__(32)
void ssim_wmma_kernel(const float* __restrict__ x1, const float* __restrict__ x2,
                      float* __restrict__ partials, int H, int W, int last) {
    // 5 field planes, padded to 32 rows x 32 cols f16
    // (rows 26..31 and cols 26..31 zeroed -> no divergence in WMMA feeding)
    __shared__ __align__(16) _Float16 sP[5][32 * 32];
    // horizontal-pass result, COLUMN-major [col][row], 16 cols x 32 rows
    __shared__ __align__(16) _Float16 sH[16 * 32];

    const int lane = threadIdx.x;
    const int ntx  = W >> 4;
    const int tx   = blockIdx.x % ntx;
    const int ty   = blockIdx.x / ntx;
    const size_t imgOff = (size_t)blockIdx.y * H * W;
    const float* __restrict__ p1 = x1 + imgOff;
    const float* __restrict__ p2 = x2 + imgOff;
    const int gr0 = ty * 16 - 5, gc0 = tx * 16 - 5;

    // ---- zero the 6 pad rows of each plane with vector stores (rows 26..31,
    //      contiguous 192 halfs per plane; 24 lanes x b128 cover it) ----
    if (lane < 24) {
        const v8h zero8 = {};
#pragma unroll
        for (int f = 0; f < 5; ++f)
            *(v8h*)&sP[f][26 * 32 + lane * 8] = zero8;
    }

    // ---- stage patch -> five f16 field planes (cols >=26 get zero) ----
    for (int pr = 0; pr < 26; ++pr) {
        const int pc = lane;                 // one col per lane
        const int gr = gr0 + pr, gc = gc0 + pc;
        float v1 = 0.0f, v2 = 0.0f;
        if (pc < 26 && gr >= 0 && gr < H && gc >= 0 && gc < W) {
            const size_t o = (size_t)gr * W + gc;
            v1 = p1[o];
            v2 = p2[o];
        }
        const int s = pr * 32 + pc;
        sP[0][s] = (_Float16)v1;
        sP[1][s] = (_Float16)v2;
        sP[2][s] = (_Float16)(v1 * v1);
        sP[3][s] = (_Float16)(v2 * v2);
        sP[4][s] = (_Float16)(v1 * v2);
    }
    __syncthreads();

    const int half = lane >> 4;   // K half-selector (ISA 16-bit operand layout)
    const int lm   = lane & 15;   // A: row M ; B: col N

    // Constant Toeplitz fragments (per-lane, computed once, pure VALU).
    v16h bh, av;                  // bh: horizontal B (32x16); av: vertical A (16x32)
#pragma unroll
    for (int e = 0; e < 16; ++e) {
        const int kB = ((e < 8) ? e : e + 8) + (half ? 8 : 0); // B: K 0-7/16-23 | 8-15/24-31
        bh[e] = (_Float16)toep(kB, lm);
        const int kA = (half ? 16 : 0) + e;                    // A: lane holds row, K=kb+e
        av[e] = (_Float16)toep(kA, lm);
    }

    const int coff = half ? 8 : 0;  // A-load column offset per lane half
    v8f O[5];

#pragma unroll
    for (int f = 0; f < 5; ++f) {
        // ---- horizontal pass: two 16-row chunks (rows 26..31 are zero) ----
        v8f dpair[2];
#pragma unroll
        for (int m = 0; m < 2; ++m) {
            const int arow = m * 16 + lm;
            const v8h lo = *(const v8h*)&sP[f][arow * 32 + coff];
            const v8h hi = *(const v8h*)&sP[f][arow * 32 + 16 + coff];
            v16h a;
#pragma unroll
            for (int e = 0; e < 8; ++e) { a[e] = lo[e]; a[8 + e] = hi[e]; }
            v8f z = {};
            dpair[m] = __builtin_amdgcn_wmma_f32_16x16x32_f16(
                false, a, false, bh, (short)0, z, false, false);
        }

        __syncthreads();  // previous field done reading sH
        // ---- D -> sH column-major: lane owns 8 consecutive rows of col lm
        //      per chunk -> one packed b128 store per chunk ----
#pragma unroll
        for (int m = 0; m < 2; ++m) {
            v8h hp;
#pragma unroll
            for (int r = 0; r < 8; ++r) hp[r] = (_Float16)dpair[m][r];
            // chunk m rows: m*16 + half*8 + r  (C/D layout: M = r or r+8)
            *(v8h*)&sH[lm * 32 + m * 16 + half * 8] = hp;
        }
        __syncthreads();

        // ---- vertical pass: B fragment = column lm, K=kb..kb+15 (contiguous) ----
        const v8h b0 = *(const v8h*)&sH[lm * 32 + (half ? 16 : 0)];
        const v8h b1 = *(const v8h*)&sH[lm * 32 + (half ? 16 : 0) + 8];
        v16h bv;
#pragma unroll
        for (int e = 0; e < 8; ++e) { bv[e] = b0[e]; bv[8 + e] = b1[e]; }
        v8f z = {};
        O[f] = __builtin_amdgcn_wmma_f32_16x16x32_f16(
            false, av, false, bv, (short)0, z, false, false);
    }

    // ---- elementwise SSIM/CS on the 8 accumulator elements per lane ----
    const float C1 = 1e-4f, C2 = 9e-4f;
    float acc = 0.0f;
#pragma unroll
    for (int r = 0; r < 8; ++r) {
        const float mu1 = O[0][r], mu2 = O[1][r];
        const float mu1s = mu1 * mu1, mu2s = mu2 * mu2, mu12 = mu1 * mu2;
        const float s1  = O[2][r] - mu1s;
        const float s2  = O[3][r] - mu2s;
        const float s12 = O[4][r] - mu12;
        const float cs  = (2.0f * s12 + C2) / (s1 + s2 + C2);
        acc += last ? ((2.0f * mu12 + C1) / (mu1s + mu2s + C1)) * cs : cs;
    }
    // wave32 butterfly reduction (fixed pattern -> deterministic)
#pragma unroll
    for (int off = 16; off > 0; off >>= 1) acc += __shfl_xor(acc, off, 32);
    if (lane == 0)
        partials[(size_t)blockIdx.y * gridDim.x + blockIdx.x] = acc;
}

// Fixed-order per-batch reduction of tile partials (no float atomics).
__global__ __launch_bounds__(32)
void reduce_partials_kernel(const float* __restrict__ partials,
                            float* __restrict__ accum_row, int nPerB) {
    const int b = blockIdx.x, lane = threadIdx.x;
    const float* p = partials + (size_t)b * nPerB;
    float s = 0.0f;
    for (int i = lane; i < nPerB; i += 32) s += p[i];
#pragma unroll
    for (int off = 16; off > 0; off >>= 1) s += __shfl_xor(s, off, 32);
    if (lane == 0) accum_row[b] = s;
}

// 2x2 average pool (all 48 channel-images of one level).
__global__ void avgpool2_kernel(const float* __restrict__ in,
                                float* __restrict__ out, int H2, int W2) {
    const int i = blockIdx.x * blockDim.x + threadIdx.x;
    const int total = 48 * H2 * W2;
    if (i >= total) return;
    const int w  = i % W2;
    const int t  = i / W2;
    const int h  = t % H2;
    const int bc = t / H2;
    const int W = W2 * 2;
    const float* p = in + ((size_t)bc * (H2 * 2) + 2 * h) * W + 2 * w;
    out[i] = 0.25f * (p[0] + p[1] + p[W] + p[W + 1]);
}

// Final: per-batch weighted product of clipped level means, then batch mean.
__global__ __launch_bounds__(32)
void msssim_final_kernel(const float* __restrict__ accum, float* __restrict__ out) {
    const int lane = threadIdx.x;
    float v = 0.0f;
    if (lane < 16) {
        const float denom[5] = {786432.0f, 196608.0f, 49152.0f, 12288.0f, 3072.0f};
        const float wgt[5]   = {0.0448f, 0.2856f, 0.3001f, 0.2363f, 0.1333f};
        float ms = 1.0f;
#pragma unroll
        for (int l = 0; l < 5; ++l) {
            float mean = accum[l * 16 + lane] / denom[l];
            mean = fmaxf(mean, 0.0f);
            ms *= powf(mean, wgt[l]);
        }
        v = ms;
    }
#pragma unroll
    for (int off = 16; off > 0; off >>= 1) v += __shfl_xor(v, off, 32);
    if (lane == 0) out[0] = v * (1.0f / 16.0f);
}

// ---------------------------------------------------------------------------
extern "C" void kernel_launch(void* const* d_in, const int* in_sizes, int n_in,
                              void* d_out, int out_size, void* d_ws, size_t ws_size,
                              hipStream_t stream) {
    (void)in_sizes; (void)n_in; (void)out_size; (void)ws_size;
    const float* img1 = (const float*)d_in[0];
    const float* img2 = (const float*)d_in[1];
    float* out = (float*)d_out;

    // workspace layout (floats): [accum 5x16 pad128][partials 49152][pyramid ~8.36M]
    float* ws       = (float*)d_ws;
    float* accum    = ws;             // 80 used, padded to 128
    float* partials = ws + 128;       // max tiles = 48*32*32 = 49152
    float* buf      = partials + 49152;

    const int Hs[5] = {512, 256, 128, 64, 32};
    const float* x1[5];
    const float* x2[5];
    float* m1[5] = {nullptr};
    float* m2[5] = {nullptr};
    x1[0] = img1; x2[0] = img2;
    size_t off = 0;
    for (int l = 1; l < 5; ++l) {
        const size_t n = (size_t)48 * Hs[l] * Hs[l];
        m1[l] = buf + off; off += n;
        m2[l] = buf + off; off += n;
        x1[l] = m1[l]; x2[l] = m2[l];
    }

    for (int l = 0; l < 5; ++l) {
        const int H = Hs[l], W = Hs[l];
        const int ntile = (H / 16) * (W / 16);
        dim3 grid(ntile, 48);
        ssim_wmma_kernel<<<grid, 32, 0, stream>>>(x1[l], x2[l], partials, H, W,
                                                  (l == 4) ? 1 : 0);
        reduce_partials_kernel<<<16, 32, 0, stream>>>(partials, accum + l * 16,
                                                      3 * ntile);
        if (l < 4) {
            const int H2 = Hs[l + 1];
            const int total = 48 * H2 * H2;
            const int blocks = (total + 255) / 256;
            avgpool2_kernel<<<blocks, 256, 0, stream>>>(x1[l], m1[l + 1], H2, H2);
            avgpool2_kernel<<<blocks, 256, 0, stream>>>(x2[l], m2[l + 1], H2, H2);
        }
    }
    msssim_final_kernel<<<1, 32, 0, stream>>>(accum, out);
}
